// RGBAdapter_67843303408304
// MI455X (gfx1250) — matrix operand order repaired and checked by reading, not compiled, not run on hardware
//
#include <hip/hip_runtime.h>
#include <hip/hip_bf16.h>
#include <math.h>

typedef __attribute__((ext_vector_type(16))) __bf16 v16bf;
typedef __attribute__((ext_vector_type(8)))  float  v8f;

#define NPIX   32768     // B*H*W
#define DIMC   768
#define MIDC   96
#define HDIM   64
#define WDIM   64
#define BATCH  8

union AFrag { __bf16 h[16]; v16bf v; };
union CFrag { float  f[8];  v8f   v; };

// load 16 floats at p[kb + {0..7, 16..23}] (A-fragment K pattern for one lane)
__device__ __forceinline__ void load16(const float* __restrict__ p, int kb, float* v) {
    const float4 a0 = *(const float4*)(p + kb + 0);
    const float4 a1 = *(const float4*)(p + kb + 4);
    const float4 a2 = *(const float4*)(p + kb + 16);
    const float4 a3 = *(const float4*)(p + kb + 20);
    v[0]=a0.x; v[1]=a0.y; v[2]=a0.z; v[3]=a0.w;
    v[4]=a1.x; v[5]=a1.y; v[6]=a1.z; v[7]=a1.w;
    v[8]=a2.x; v[9]=a2.y; v[10]=a2.z; v[11]=a2.w;
    v[12]=a3.x; v[13]=a3.y; v[14]=a3.z; v[15]=a3.w;
}

// ---------------------------------------------------------------------------
// Pack weight W[N,K] (f32, row-major) into WMMA B-fragment order (bf16).
// Index: (((kk*NT + nt)*32 + lane)*16 + e);  k = kk*32 + (lane>=16?8:0) + (e&7) + (e>=8?16:0),
// n = nt*16 + lane%16.  (B 32x16 layout: lanes0-15 K=0..7,16..23, lanes16-31 K=8..15,24..31)
// ---------------------------------------------------------------------------
__global__ void k_pack(const float* __restrict__ W, __bf16* __restrict__ out,
                       int N, int Kdim) {
    int idx = blockIdx.x * 256 + threadIdx.x;
    int total = (Kdim >> 5) * (N >> 4) * 512;
    if (idx >= total) return;
    int e    = idx & 15;
    int lane = (idx >> 4) & 31;
    int nt   = (idx >> 9) % (N >> 4);
    int kk   = idx / (512 * (N >> 4));
    int k = kk * 32 + ((lane & 16) ? 8 : 0) + (e & 7) + ((e & 8) ? 16 : 0);
    int n = nt * 16 + (lane & 15);
    out[idx] = (__bf16)W[(size_t)n * Kdim + k];
}

// ---------------------------------------------------------------------------
// Fused LayerNorm(768) + Linear(768->96), output NHWC [NPIX, 96] f32.
// One wave = 16 pixels x 96 outputs. 24 K-steps x 6 N-tiles of wmma bf16.
// ---------------------------------------------------------------------------
__global__ void k_ln_down(const float* __restrict__ x,
                          const float* __restrict__ lnw, const float* __restrict__ lnb,
                          const __bf16* __restrict__ wpack,
                          const float* __restrict__ bias,
                          float* __restrict__ out) {
    int tile = blockIdx.x * 8 + (threadIdx.x >> 5);   // 2048 tiles of 16 pixels
    int lane = threadIdx.x & 31;
    int m = lane & 15;
    int khalf = (lane & 16) ? 8 : 0;                  // A-layout K half
    const float* row = x + (size_t)(tile * 16 + m) * DIMC;

    // pass 1: LN stats (each lane covers half the row; pair via shfl_xor 16)
    float s = 0.f, ss = 0.f;
    for (int kk = 0; kk < 24; ++kk) {
        float va[16];
        load16(row, kk * 32 + khalf, va);
#pragma unroll
        for (int e = 0; e < 16; ++e) { s += va[e]; ss += va[e] * va[e]; }
    }
    s  += __shfl_xor(s, 16, 32);
    ss += __shfl_xor(ss, 16, 32);
    float mean = s * (1.f / DIMC);
    float var  = ss * (1.f / DIMC) - mean * mean;
    float rstd = rsqrtf(var + 1e-6f);

    v8f acc[6];
#pragma unroll
    for (int i = 0; i < 6; ++i) acc[i] = (v8f){0,0,0,0,0,0,0,0};

    const v16bf* bp = (const v16bf*)wpack;
    for (int kk = 0; kk < 24; ++kk) {
        int kb = kk * 32 + khalf;
        float va[16], wv[16], bv[16];
        load16(row, kb, va);
        load16(lnw, kb, wv);
        load16(lnb, kb, bv);
        AFrag a;
#pragma unroll
        for (int e = 0; e < 16; ++e)
            a.h[e] = (__bf16)((va[e] - mean) * rstd * wv[e] + bv[e]);
#pragma unroll
        for (int nt = 0; nt < 6; ++nt) {
            v16bf bf = bp[(kk * 6 + nt) * 32 + lane];
            acc[nt] = __builtin_amdgcn_wmma_f32_16x16x32_bf16(
                false, a.v, false, bf, (short)0, acc[nt], false, false);
        }
    }

    int n0 = lane & 15;
    int mrow = (lane & 16) ? 8 : 0;
#pragma unroll
    for (int nt = 0; nt < 6; ++nt) {
        int n = nt * 16 + n0;
        float bb = bias[n];
        CFrag c; c.v = acc[nt];
#pragma unroll
        for (int v = 0; v < 8; ++v) {
            int pix = tile * 16 + v + mrow;
            out[(size_t)pix * MIDC + n] = c.f[v] + bb;
        }
    }
}

// ---------------------------------------------------------------------------
// Depthwise 3x3 (zero pad) + bias + SiLU, NHWC.
// ---------------------------------------------------------------------------
__global__ void k_dwconv(const float* __restrict__ temb,
                         const float* __restrict__ dww, const float* __restrict__ dwb,
                         float* __restrict__ y) {
    size_t idx = (size_t)blockIdx.x * 256 + threadIdx.x;      // NPIX*96
    int c = (int)(idx % MIDC);
    int p = (int)(idx / MIDC);
    int b = p >> 12, ij = p & 4095, i = ij >> 6, j = ij & 63;
    float a = dwb[c];
#pragma unroll
    for (int di = 0; di < 3; ++di) {
        int ii = i + di - 1;
        if (ii < 0 || ii >= HDIM) continue;
#pragma unroll
        for (int dj = 0; dj < 3; ++dj) {
            int jj = j + dj - 1;
            if (jj < 0 || jj >= WDIM) continue;
            a += temb[((size_t)((b << 12) + (ii << 6) + jj)) * MIDC + c] *
                 dww[c * 9 + di * 3 + dj];
        }
    }
    y[idx] = a / (1.f + __expf(-a));
}

// ---------------------------------------------------------------------------
// Style GEMM [NPIX,96] x [96,192] via wmma + fused FiLM:
// common = x_emb * (1 + gamma) + beta
// ---------------------------------------------------------------------------
__global__ void k_style_common(const float* __restrict__ ybuf,
                               const __bf16* __restrict__ ppw,
                               const float* __restrict__ pwb,
                               const float* __restrict__ xemb,
                               float* __restrict__ common) {
    int tile = blockIdx.x * 8 + (threadIdx.x >> 5);
    int lane = threadIdx.x & 31;
    int m = lane & 15;
    int khalf = (lane & 16) ? 8 : 0;
    const float* row = ybuf + (size_t)(tile * 16 + m) * MIDC;

    v8f acc[12];
#pragma unroll
    for (int i = 0; i < 12; ++i) acc[i] = (v8f){0,0,0,0,0,0,0,0};

    const v16bf* bp = (const v16bf*)ppw;
#pragma unroll
    for (int kk = 0; kk < 3; ++kk) {
        float va[16];
        load16(row, kk * 32 + khalf, va);
        AFrag a;
#pragma unroll
        for (int e = 0; e < 16; ++e) a.h[e] = (__bf16)va[e];
#pragma unroll
        for (int nt = 0; nt < 12; ++nt) {
            v16bf bf = bp[(kk * 12 + nt) * 32 + lane];
            acc[nt] = __builtin_amdgcn_wmma_f32_16x16x32_bf16(
                false, a.v, false, bf, (short)0, acc[nt], false, false);
        }
    }

    int n0 = lane & 15;
    int mrow = (lane & 16) ? 8 : 0;
#pragma unroll
    for (int nt = 0; nt < 6; ++nt) {
        int cg = nt * 16 + n0;
        float gb = pwb[cg], bb = pwb[MIDC + cg];
        CFrag g, bt; g.v = acc[nt]; bt.v = acc[nt + 6];
#pragma unroll
        for (int v = 0; v < 8; ++v) {
            int pix = tile * 16 + v + mrow;
            float xe = xemb[(size_t)pix * MIDC + cg];
            common[(size_t)pix * MIDC + cg] =
                xe * (1.f + g.f[v] + gb) + (bt.f[v] + bb);
        }
    }
}

// ---------------------------------------------------------------------------
// gvec[b][c] = mean over 4096 pixels of x_emb (LDS partial sums, no globals)
// ---------------------------------------------------------------------------
__global__ void k_gvec(const float* __restrict__ xemb, float* __restrict__ gvec) {
    __shared__ float sums[MIDC];
    int b = blockIdx.x;
    int tid = threadIdx.x;
    if (tid < MIDC) sums[tid] = 0.f;
    __syncthreads();
    const float* base = xemb + (size_t)b * 4096 * MIDC;
    // stride 256: channel cycles with period 3 (3*256 = 8*96)
    float l0 = 0.f, l1 = 0.f, l2 = 0.f;
    for (int f = tid; f < 4096 * MIDC; f += 768) {
        l0 += base[f]; l1 += base[f + 256]; l2 += base[f + 512];
    }
    int c0 = tid % MIDC;
    atomicAdd(&sums[c0], l0);
    atomicAdd(&sums[(c0 + 64) % MIDC], l1);
    atomicAdd(&sums[(c0 + 32) % MIDC], l2);
    __syncthreads();
    if (tid < MIDC) gvec[b * MIDC + tid] = sums[tid] * (1.f / 4096.f);
}

// ---------------------------------------------------------------------------
// MLP: relu(gvec@mlp1^T) -> softmax(.@mlp2^T). One warp per batch.
// ---------------------------------------------------------------------------
__global__ void k_mlp(const float* __restrict__ gvec,
                      const float* __restrict__ w1, const float* __restrict__ b1,
                      const float* __restrict__ w2, const float* __restrict__ b2,
                      float* __restrict__ wbuf) {
    __shared__ float hid[BATCH][24];
    __shared__ float lg[BATCH][4];
    int b = threadIdx.x >> 5;
    int lane = threadIdx.x & 31;
    if (lane < 24) {
        float a = b1[lane];
        for (int c = 0; c < MIDC; ++c) a += gvec[b * MIDC + c] * w1[lane * MIDC + c];
        hid[b][lane] = fmaxf(a, 0.f);
    }
    __syncthreads();
    if (lane < 4) {
        float a = b2[lane];
        for (int j = 0; j < 24; ++j) a += hid[b][j] * w2[lane * 24 + j];
        lg[b][lane] = a;
    }
    __syncthreads();
    if (lane < 4) {
        float mx = fmaxf(fmaxf(lg[b][0], lg[b][1]), fmaxf(lg[b][2], lg[b][3]));
        float den = 0.f;
        for (int k = 0; k < 4; ++k) den += __expf(lg[b][k] - mx);
        wbuf[b * 4 + lane] = __expf(lg[b][lane] - mx) / den;
    }
}

// ---------------------------------------------------------------------------
// dyn_t[b][i][j][c] = sum_k wbuf[b,k] * basis[k,c,i,j]  (channel-last)
// ---------------------------------------------------------------------------
__global__ void k_dyn(const float* __restrict__ wbuf, const float* __restrict__ basis,
                      float* __restrict__ dynt) {
    int idx = blockIdx.x * 256 + threadIdx.x;
    if (idx >= BATCH * 49 * MIDC) return;
    int c = idx % MIDC;
    int r = idx / MIDC;
    int j = r % 7, i = (r / 7) % 7, b = r / 49;
    float a = 0.f;
#pragma unroll
    for (int k = 0; k < 4; ++k)
        a += wbuf[b * 4 + k] * basis[((k * MIDC + c) * 7 + i) * 7 + j];
    dynt[idx] = a;
}

// ---------------------------------------------------------------------------
// Circular 7x7 conv (== rfft2/irfft2 of rolled padded kernel):
// out[i,j] = sum_{u,v=-3..3} dyn[u+3,v+3] * common[(i-u)&63, (j-v)&63]
// ---------------------------------------------------------------------------
__global__ void k_circ(const float* __restrict__ common, const float* __restrict__ dynt,
                       float* __restrict__ convout) {
    size_t idx = (size_t)blockIdx.x * 256 + threadIdx.x;
    int c = (int)(idx % MIDC);
    int p = (int)(idx / MIDC);
    int b = p >> 12, i = (p >> 6) & 63, j = p & 63;
    float k49[49];
    const float* dp = dynt + (size_t)b * 49 * MIDC + c;
#pragma unroll
    for (int t = 0; t < 49; ++t) k49[t] = dp[t * MIDC];
    const float* cb = common + ((size_t)b << 12) * MIDC;
    float a = 0.f;
#pragma unroll
    for (int u = 0; u < 7; ++u) {
        int ii = (i - (u - 3)) & 63;
#pragma unroll
        for (int v = 0; v < 7; ++v) {
            int jj = (j - (v - 3)) & 63;
            a += k49[u * 7 + v] * cb[(size_t)((ii << 6) + jj) * MIDC + c];
        }
    }
    convout[idx] = a;
}

// ---------------------------------------------------------------------------
// Up GEMM [NPIX,96]x[96,768] + up_b + residual x. Block = 1 pixel tile,
// 8 waves = 8 N-groups of 96 outputs each.
// ---------------------------------------------------------------------------
__global__ void k_up(const float* __restrict__ conv, const __bf16* __restrict__ pup,
                     const float* __restrict__ upb, const float* __restrict__ x,
                     float* __restrict__ out) {
    int tile = blockIdx.x;                 // 2048
    int ng = threadIdx.x >> 5;             // 0..7
    int lane = threadIdx.x & 31;
    int m = lane & 15;
    int khalf = (lane & 16) ? 8 : 0;
    const float* row = conv + (size_t)(tile * 16 + m) * MIDC;

    v8f acc[6];
#pragma unroll
    for (int i = 0; i < 6; ++i) acc[i] = (v8f){0,0,0,0,0,0,0,0};

    const v16bf* bp = (const v16bf*)pup;
#pragma unroll
    for (int kk = 0; kk < 3; ++kk) {
        float va[16];
        load16(row, kk * 32 + khalf, va);
        AFrag a;
#pragma unroll
        for (int e = 0; e < 16; ++e) a.h[e] = (__bf16)va[e];
#pragma unroll
        for (int nt6 = 0; nt6 < 6; ++nt6) {
            int ntg = ng * 6 + nt6;        // 0..47
            v16bf bf = bp[(kk * 48 + ntg) * 32 + lane];
            acc[nt6] = __builtin_amdgcn_wmma_f32_16x16x32_bf16(
                false, a.v, false, bf, (short)0, acc[nt6], false, false);
        }
    }

    int n0 = lane & 15;
    int mrow = (lane & 16) ? 8 : 0;
#pragma unroll
    for (int nt6 = 0; nt6 < 6; ++nt6) {
        int n = ng * MIDC + nt6 * 16 + n0;
        float bb = upb[n];
        CFrag c; c.v = acc[nt6];
#pragma unroll
        for (int v = 0; v < 8; ++v) {
            size_t pix = (size_t)(tile * 16 + v + mrow);
            out[pix * DIMC + n] = c.f[v] + bb + x[pix * DIMC + n];
        }
    }
}

// ---------------------------------------------------------------------------
extern "C" void kernel_launch(void* const* d_in, const int* in_sizes, int n_in,
                              void* d_out, int out_size, void* d_ws, size_t ws_size,
                              hipStream_t stream) {
    (void)in_sizes; (void)n_in; (void)out_size; (void)ws_size;
    const float* x      = (const float*)d_in[0];
    const float* t      = (const float*)d_in[1];
    const float* ln_w   = (const float*)d_in[2];
    const float* ln_b   = (const float*)d_in[3];
    const float* down_w = (const float*)d_in[4];
    const float* down_b = (const float*)d_in[5];
    const float* ln1_w  = (const float*)d_in[6];
    const float* ln1_b  = (const float*)d_in[7];
    const float* down1_w= (const float*)d_in[8];
    const float* down1_b= (const float*)d_in[9];
    const float* dw_w   = (const float*)d_in[10];
    const float* dw_b   = (const float*)d_in[11];
    const float* pw_w   = (const float*)d_in[12];
    const float* pw_b   = (const float*)d_in[13];
    const float* mlp1_w = (const float*)d_in[14];
    const float* mlp1_b = (const float*)d_in[15];
    const float* mlp2_w = (const float*)d_in[16];
    const float* mlp2_b = (const float*)d_in[17];
    const float* basis  = (const float*)d_in[18];
    const float* up_w   = (const float*)d_in[19];
    const float* up_b   = (const float*)d_in[20];
    float* out = (float*)d_out;

    // workspace carving (all 256B aligned)
    char* ws = (char*)d_ws;
    size_t off = 0;
    auto carve = [&](size_t bytes) { char* p = ws + off; off += (bytes + 255) & ~(size_t)255; return p; };
    float*  x_emb   = (float*)carve((size_t)NPIX * MIDC * 4);
    float*  t_emb   = (float*)carve((size_t)NPIX * MIDC * 4);
    float*  ybuf    = (float*)carve((size_t)NPIX * MIDC * 4);
    float*  common  = (float*)carve((size_t)NPIX * MIDC * 4);
    float*  convout = (float*)carve((size_t)NPIX * MIDC * 4);
    __bf16* pdown   = (__bf16*)carve((size_t)24 * 6 * 512 * 2);
    __bf16* pdown1  = (__bf16*)carve((size_t)24 * 6 * 512 * 2);
    __bf16* ppw     = (__bf16*)carve((size_t)3 * 12 * 512 * 2);
    __bf16* pup     = (__bf16*)carve((size_t)3 * 48 * 512 * 2);
    float*  gvec    = (float*)carve((size_t)BATCH * MIDC * 4);
    float*  wbuf    = (float*)carve((size_t)BATCH * 4 * 4);
    float*  dynt    = (float*)carve((size_t)BATCH * 49 * MIDC * 4);

    // 1) pack weights into WMMA B-fragment order (bf16)
    k_pack<<<288, 256, 0, stream>>>(down_w,  pdown,  MIDC,     DIMC);
    k_pack<<<288, 256, 0, stream>>>(down1_w, pdown1, MIDC,     DIMC);
    k_pack<<<72,  256, 0, stream>>>(pw_w,    ppw,    2 * MIDC, MIDC);
    k_pack<<<288, 256, 0, stream>>>(up_w,    pup,    DIMC,     MIDC);

    // 2) fused LN + down projections
    k_ln_down<<<256, 256, 0, stream>>>(x, ln_w,  ln_b,  pdown,  down_b,  x_emb);
    k_ln_down<<<256, 256, 0, stream>>>(t, ln1_w, ln1_b, pdown1, down1_b, t_emb);

    // 3) depthwise conv + SiLU
    k_dwconv<<<12288, 256, 0, stream>>>(t_emb, dw_w, dw_b, ybuf);

    // 4) style GEMM + FiLM -> common
    k_style_common<<<256, 256, 0, stream>>>(ybuf, ppw, pw_b, x_emb, common);

    // 5) pooled guidance vector, MLP+softmax, dynamic kernel
    k_gvec<<<BATCH, 256, 0, stream>>>(x_emb, gvec);
    k_mlp<<<1, 256, 0, stream>>>(gvec, mlp1_w, mlp1_b, mlp2_w, mlp2_b, wbuf);
    k_dyn<<<147, 256, 0, stream>>>(wbuf, basis, dynt);

    // 6) circular 7x7 convolution (replaces rfft2/irfft2)
    k_circ<<<12288, 256, 0, stream>>>(common, dynt, convout);

    // 7) up projection + bias + residual
    k_up<<<2048, 256, 0, stream>>>(convout, pup, up_b, x, out);
}
